// MultiHeadAttention_2886218023112
// MI455X (gfx1250) — matrix operand order repaired and checked by reading, not compiled, not run on hardware
//
#include <hip/hip_runtime.h>

// ---------------------------------------------------------------------------
// MHA forward for MI455X (gfx1250, wave32, WMMA).
//  B=4, S=2048, D=512, H=8, dk=64.
//  Pipeline: f32->f16 cvt | QKV proj (WMMA, pipelined) |
//            one-pass flash attention (WMMA, per (b,h,qtile) wave) |
//            head-mean kernel (WMMA recompute, register acc, NT stores) |
//            output proj (WMMA, f32 out).
// ---------------------------------------------------------------------------

typedef __attribute__((ext_vector_type(16))) _Float16 v16h;
typedef __attribute__((ext_vector_type(8)))  float    v8f;

#define D_MODEL 512
#define NHEAD   8
#define DK      64
#define SEQ     2048
#define BATCH   4

// A-fragment (16x32 f16, M x K), row-major source with leading dim ld.
// lanes 0-15 hold M=lane, K=0..15; lanes 16-31 M=lane-16,
// K=16..31 (same contiguous K order as load_fragB).
__device__ __forceinline__ v16h load_fragA(const _Float16* p, int ld) {
  const int row = threadIdx.x & 15;
  const int kg  = (threadIdx.x >> 4) & 1;
  const _Float16* rp = p + (size_t)row * ld + kg * 16;
  v16h f;
#pragma unroll
  for (int j = 0; j < 16; ++j) f[j] = rp[j];
  return f;
}

// B-fragment (32x16 f16, K x N). Memory holds B^T row-major: Bt[n][k].
// Lane holds column N=lane&15; lanes 0-15 K=0..15, lanes 16-31 K=16..31 ->
// one contiguous 32B load per lane.
__device__ __forceinline__ v16h load_fragB(const _Float16* p, int ld) {
  const _Float16* rp = p + (size_t)(threadIdx.x & 15) * ld +
                       ((threadIdx.x >> 4) & 1) * 16;
  v16h f;
#pragma unroll
  for (int j = 0; j < 16; ++j) f[j] = rp[j];
  return f;
}

__device__ __forceinline__ v8f wmma_f16(v16h a, v16h b, v8f c) {
  return __builtin_amdgcn_wmma_f32_16x16x32_f16(false, a, false, b,
                                                (short)0, c, false, false);
}

// ---------------------------------------------------------------------------
__global__ void cvt_f32_f16(const float* __restrict__ src,
                            _Float16* __restrict__ dst, int n) {
  int i = (blockIdx.x * blockDim.x + threadIdx.x) * 4;
  if (i < n) {
    dst[i + 0] = (_Float16)src[i + 0];
    dst[i + 1] = (_Float16)src[i + 1];
    dst[i + 2] = (_Float16)src[i + 2];
    dst[i + 3] = (_Float16)src[i + 3];
  }
}

// ---------------------------------------------------------------------------
// QKV projection: P = X (8192x512) @ W^T + b. One wave computes a 16x64 tile.
// Software-pipelined over K: next fragments are loaded before current WMMAs'
// results are needed, so s_wait_loadcnt overlaps with 4 WMMAs.
// vmode 0: head-major store dst[b][h][s][dk]   (Q, K)
// vmode 1: transposed store dst[b][h][dk][s]   (V -> V^T for PV B-fragments)
__global__ __launch_bounds__(32)
void proj_qkv_kernel(const _Float16* __restrict__ X,
                     const _Float16* __restrict__ W,
                     const float* __restrict__ bias,
                     _Float16* __restrict__ dst, int vmode) {
  const int r0 = blockIdx.x * 16;
  const int c0 = blockIdx.y * 64;
  const _Float16* Xp = X + (size_t)r0 * D_MODEL;
  const _Float16* W0 = W + (size_t)(c0 +  0) * D_MODEL;
  const _Float16* W1 = W + (size_t)(c0 + 16) * D_MODEL;
  const _Float16* W2 = W + (size_t)(c0 + 32) * D_MODEL;
  const _Float16* W3 = W + (size_t)(c0 + 48) * D_MODEL;

  v8f acc0 = {}, acc1 = {}, acc2 = {}, acc3 = {};
  v16h a  = load_fragA(Xp, D_MODEL);
  v16h b0 = load_fragB(W0, D_MODEL), b1 = load_fragB(W1, D_MODEL);
  v16h b2 = load_fragB(W2, D_MODEL), b3 = load_fragB(W3, D_MODEL);
  for (int kc = 32; kc < D_MODEL; kc += 32) {
    v16h an  = load_fragA(Xp + kc, D_MODEL);
    v16h bn0 = load_fragB(W0 + kc, D_MODEL), bn1 = load_fragB(W1 + kc, D_MODEL);
    v16h bn2 = load_fragB(W2 + kc, D_MODEL), bn3 = load_fragB(W3 + kc, D_MODEL);
    acc0 = wmma_f16(a, b0, acc0);
    acc1 = wmma_f16(a, b1, acc1);
    acc2 = wmma_f16(a, b2, acc2);
    acc3 = wmma_f16(a, b3, acc3);
    a = an; b0 = bn0; b1 = bn1; b2 = bn2; b3 = bn3;
  }
  acc0 = wmma_f16(a, b0, acc0);
  acc1 = wmma_f16(a, b1, acc1);
  acc2 = wmma_f16(a, b2, acc2);
  acc3 = wmma_f16(a, b3, acc3);

  v8f acc[4] = {acc0, acc1, acc2, acc3};
  const int b  = r0 >> 11;
  const int s0 = r0 & (SEQ - 1);
  const int h  = c0 >> 6;
  const int n  = threadIdx.x & 15;
  const int hi = (threadIdx.x >> 4) & 1;
#pragma unroll
  for (int t = 0; t < 4; ++t) {
#pragma unroll
    for (int r = 0; r < 8; ++r) {
      const int m   = r + 8 * hi;
      const int col = c0 + t * 16 + n;
      const int dkc = col & (DK - 1);
      const float v = acc[t][r] + bias[col];
      if (vmode == 0)
        dst[((size_t)(b * NHEAD + h) * SEQ + (s0 + m)) * DK + dkc] = (_Float16)v;
      else
        dst[((size_t)(b * NHEAD + h) * DK + dkc) * SEQ + (s0 + m)] = (_Float16)v;
    }
  }
}

// ---------------------------------------------------------------------------
// One-pass flash attention. grid (S/16, H, B), 1 wave per block.
// Each wave: 16 query rows of one (b,h); online softmax over 32-key chunks;
// emits f16 context slice and per-row (m, l) stats for the mean kernel.
__global__ __launch_bounds__(32)
void flash_attn_kernel(const _Float16* __restrict__ Qh,
                       const _Float16* __restrict__ Kh,
                       const _Float16* __restrict__ Vt,
                       const unsigned char* __restrict__ mask,
                       _Float16* __restrict__ ctx, float* __restrict__ ml) {
  __shared__ _Float16 pst[16 * 32];    // C-layout -> A-layout prob restage
  const int q0 = blockIdx.x * 16;
  const int h  = blockIdx.y;
  const int b  = blockIdx.z;
  const int n  = threadIdx.x & 15;
  const int hi = (threadIdx.x >> 4) & 1;

  const _Float16* Qp = Qh + ((size_t)(b * NHEAD + h) * SEQ + q0) * DK;
  const _Float16* Kp = Kh + (size_t)(b * NHEAD + h) * SEQ * DK;
  const _Float16* Vp = Vt + (size_t)(b * NHEAD + h) * DK * SEQ;
  const unsigned char* mrow = mask + (size_t)b * SEQ;

  const v16h aq0 = load_fragA(Qp, DK);
  const v16h aq1 = load_fragA(Qp + 32, DK);

  float mrun[8], lrun[8];
#pragma unroll
  for (int r = 0; r < 8; ++r) { mrun[r] = -1e30f; lrun[r] = 0.0f; }
  v8f o0 = {}, o1 = {}, o2 = {}, o3 = {};

  // preload K fragments for chunk 0
  v16h k00 = load_fragB(Kp, DK);
  v16h k01 = load_fragB(Kp + 32, DK);
  v16h k10 = load_fragB(Kp + (size_t)16 * DK, DK);
  v16h k11 = load_fragB(Kp + (size_t)16 * DK + 32, DK);

  for (int kc = 0; kc < SEQ; kc += 32) {
    // V fragments for this chunk: issue early, consumed after softmax VALU
    v16h vb0 = load_fragB(Vp + (size_t)( 0) * SEQ + kc, SEQ);
    v16h vb1 = load_fragB(Vp + (size_t)(16) * SEQ + kc, SEQ);
    v16h vb2 = load_fragB(Vp + (size_t)(32) * SEQ + kc, SEQ);
    v16h vb3 = load_fragB(Vp + (size_t)(48) * SEQ + kc, SEQ);

    v8f s0v = {}, s1v = {};
    s0v = wmma_f16(aq0, k00, s0v);
    s0v = wmma_f16(aq1, k01, s0v);
    s1v = wmma_f16(aq0, k10, s1v);
    s1v = wmma_f16(aq1, k11, s1v);

    // preload next chunk's K fragments (wrap on last iter; dead but harmless)
    const int kn = (kc + 32) & (SEQ - 1);
    k00 = load_fragB(Kp + (size_t)kn * DK, DK);
    k01 = load_fragB(Kp + (size_t)kn * DK + 32, DK);
    k10 = load_fragB(Kp + (size_t)(kn + 16) * DK, DK);
    k11 = load_fragB(Kp + (size_t)(kn + 16) * DK + 32, DK);

    const bool v0 = mrow[kc + n] != 0;
    const bool v1 = mrow[kc + 16 + n] != 0;

    float scale[8];
#pragma unroll
    for (int r = 0; r < 8; ++r) {
      const int m = r + 8 * hi;
      const float a0 = s0v[r] * 0.125f;          // 1/sqrt(dk)
      const float a1 = s1v[r] * 0.125f;
      float cm = fmaxf(v0 ? a0 : -1e30f, v1 ? a1 : -1e30f);
#pragma unroll
      for (int d = 1; d < 16; d <<= 1) cm = fmaxf(cm, __shfl_xor(cm, d, 32));
      const float mn = fmaxf(mrun[r], cm);
      const float p0 = v0 ? __expf(a0 - mn) : 0.0f;   // unnormalized
      const float p1 = v1 ? __expf(a1 - mn) : 0.0f;
      pst[m * 32 + n]      = (_Float16)p0;
      pst[m * 32 + 16 + n] = (_Float16)p1;
      float ps = p0 + p1;
#pragma unroll
      for (int d = 1; d < 16; d <<= 1) ps += __shfl_xor(ps, d, 32);
      scale[r] = __expf(mrun[r] - mn);           // ==1 if mn unchanged
      lrun[r] = lrun[r] * scale[r] + ps;
      mrun[r] = mn;
    }
    // rescale running output, then accumulate P@V
#pragma unroll
    for (int r = 0; r < 8; ++r) {
      o0[r] *= scale[r]; o1[r] *= scale[r]; o2[r] *= scale[r]; o3[r] *= scale[r];
    }
    const v16h ap = load_fragA(pst, 32);
    o0 = wmma_f16(ap, vb0, o0);
    o1 = wmma_f16(ap, vb1, o1);
    o2 = wmma_f16(ap, vb2, o2);
    o3 = wmma_f16(ap, vb3, o3);
  }

  float invl[8];
#pragma unroll
  for (int r = 0; r < 8; ++r) invl[r] = lrun[r] > 0.0f ? 1.0f / lrun[r] : 0.0f;

  v8f o[4] = {o0, o1, o2, o3};
#pragma unroll
  for (int t = 0; t < 4; ++t)
#pragma unroll
    for (int r = 0; r < 8; ++r) {
      const int m = r + 8 * hi;
      ctx[((size_t)b * SEQ + q0 + m) * D_MODEL + h * DK + t * 16 + n] =
          (_Float16)(o[t][r] * invl[r]);
    }

  // one lane per 16-lane group stores (m, l) per row
  if (n == 0) {
    float2* mlp = (float2*)ml + (size_t)(b * NHEAD + h) * SEQ + q0;
#pragma unroll
    for (int r = 0; r < 8; ++r) {
      float2 st; st.x = mrun[r]; st.y = lrun[r];
      mlp[r + 8 * hi] = st;
    }
  }
}

// ---------------------------------------------------------------------------
// Head-mean of attention probabilities. grid (S/32 key chunks, S/16 q tiles,
// B), 1 wave per block. Recomputes the 16x32 score tile for all 8 heads with
// the stored (m,l) stats, accumulates the mean in registers, writes the
// (B,S,S) output once with non-temporal stores (streaming, bypass L2 keep).
__global__ __launch_bounds__(32)
void attn_mean_kernel(const _Float16* __restrict__ Qh,
                      const _Float16* __restrict__ Kh,
                      const unsigned char* __restrict__ mask,
                      const float* __restrict__ ml,
                      float* __restrict__ attn_out) {
  const int kc = blockIdx.x * 32;
  const int q0 = blockIdx.y * 16;
  const int b  = blockIdx.z;
  const int n  = threadIdx.x & 15;
  const int hi = (threadIdx.x >> 4) & 1;

  const bool v0 = mask[(size_t)b * SEQ + kc + n] != 0;
  const bool v1 = mask[(size_t)b * SEQ + kc + 16 + n] != 0;

  float macc0[8], macc1[8];
#pragma unroll
  for (int r = 0; r < 8; ++r) { macc0[r] = 0.0f; macc1[r] = 0.0f; }

  for (int h = 0; h < NHEAD; ++h) {
    const _Float16* Qp = Qh + ((size_t)(b * NHEAD + h) * SEQ + q0) * DK;
    const _Float16* Kp = Kh + (size_t)(b * NHEAD + h) * SEQ * DK;
    const v16h aq0 = load_fragA(Qp, DK);
    const v16h aq1 = load_fragA(Qp + 32, DK);
    v8f s0v = {}, s1v = {};
    s0v = wmma_f16(aq0, load_fragB(Kp + (size_t)(kc)      * DK, DK), s0v);
    s0v = wmma_f16(aq1, load_fragB(Kp + (size_t)(kc)      * DK + 32, DK), s0v);
    s1v = wmma_f16(aq0, load_fragB(Kp + (size_t)(kc + 16) * DK, DK), s1v);
    s1v = wmma_f16(aq1, load_fragB(Kp + (size_t)(kc + 16) * DK + 32, DK), s1v);

    const float2* mlp = (const float2*)ml + (size_t)(b * NHEAD + h) * SEQ + q0;
#pragma unroll
    for (int r = 0; r < 8; ++r) {
      const float2 st = mlp[r + 8 * hi];
      const float inv = st.y > 0.0f ? 1.0f / st.y : 0.0f;   // NaN fix
      macc0[r] += v0 ? __expf(s0v[r] * 0.125f - st.x) * inv : 0.0f;
      macc1[r] += v1 ? __expf(s1v[r] * 0.125f - st.x) * inv : 0.0f;
    }
  }

  float* ap = attn_out + ((size_t)b * SEQ + q0) * SEQ + kc;
#pragma unroll
  for (int r = 0; r < 8; ++r) {
    const int m = r + 8 * hi;
    __builtin_nontemporal_store(macc0[r] * (1.0f / NHEAD), ap + (size_t)m * SEQ + n);
    __builtin_nontemporal_store(macc1[r] * (1.0f / NHEAD), ap + (size_t)m * SEQ + 16 + n);
  }
}

// ---------------------------------------------------------------------------
// Output projection: out = ctx @ Wo^T + bo, f32 result straight to d_out.
__global__ __launch_bounds__(32)
void oproj_kernel(const _Float16* __restrict__ ctx, const _Float16* __restrict__ Wo,
                  const float* __restrict__ bo, float* __restrict__ out) {
  const int r0 = blockIdx.x * 16;
  const int c0 = blockIdx.y * 64;
  const _Float16* Xp = ctx + (size_t)r0 * D_MODEL;
  const _Float16* W0 = Wo + (size_t)(c0 +  0) * D_MODEL;
  const _Float16* W1 = Wo + (size_t)(c0 + 16) * D_MODEL;
  const _Float16* W2 = Wo + (size_t)(c0 + 32) * D_MODEL;
  const _Float16* W3 = Wo + (size_t)(c0 + 48) * D_MODEL;

  v8f acc0 = {}, acc1 = {}, acc2 = {}, acc3 = {};
  v16h a  = load_fragA(Xp, D_MODEL);
  v16h b0 = load_fragB(W0, D_MODEL), b1 = load_fragB(W1, D_MODEL);
  v16h b2 = load_fragB(W2, D_MODEL), b3 = load_fragB(W3, D_MODEL);
  for (int kc = 32; kc < D_MODEL; kc += 32) {
    v16h an  = load_fragA(Xp + kc, D_MODEL);
    v16h bn0 = load_fragB(W0 + kc, D_MODEL), bn1 = load_fragB(W1 + kc, D_MODEL);
    v16h bn2 = load_fragB(W2 + kc, D_MODEL), bn3 = load_fragB(W3 + kc, D_MODEL);
    acc0 = wmma_f16(a, b0, acc0);
    acc1 = wmma_f16(a, b1, acc1);
    acc2 = wmma_f16(a, b2, acc2);
    acc3 = wmma_f16(a, b3, acc3);
    a = an; b0 = bn0; b1 = bn1; b2 = bn2; b3 = bn3;
  }
  acc0 = wmma_f16(a, b0, acc0);
  acc1 = wmma_f16(a, b1, acc1);
  acc2 = wmma_f16(a, b2, acc2);
  acc3 = wmma_f16(a, b3, acc3);

  v8f acc[4] = {acc0, acc1, acc2, acc3};
  const int n = threadIdx.x & 15, hi = (threadIdx.x >> 4) & 1;
#pragma unroll
  for (int t = 0; t < 4; ++t)
#pragma unroll
    for (int r = 0; r < 8; ++r)
      out[((size_t)r0 + r + 8 * hi) * D_MODEL + c0 + t * 16 + n] =
          acc[t][r] + bo[c0 + t * 16 + n];
}

// ---------------------------------------------------------------------------
extern "C" void kernel_launch(void* const* d_in, const int* in_sizes, int n_in,
                              void* d_out, int out_size, void* d_ws, size_t ws_size,
                              hipStream_t stream) {
  const float* q  = (const float*)d_in[0];
  const float* k  = (const float*)d_in[1];
  const float* v  = (const float*)d_in[2];
  const unsigned char* mask = (const unsigned char*)d_in[3];   // jax bool = 1B
  const float* Wq = (const float*)d_in[5];
  const float* bq = (const float*)d_in[6];
  const float* Wk = (const float*)d_in[7];
  const float* bk = (const float*)d_in[8];
  const float* Wv = (const float*)d_in[9];
  const float* bv = (const float*)d_in[10];
  const float* Wo = (const float*)d_in[11];
  const float* bo = (const float*)d_in[12];
  (void)in_sizes; (void)n_in; (void)out_size; (void)ws_size;

  const size_t NX = (size_t)BATCH * SEQ * D_MODEL;   // 4,194,304
  const size_t NW = (size_t)D_MODEL * D_MODEL;       //   262,144
  _Float16* ws   = (_Float16*)d_ws;
  _Float16* xq16 = ws;
  _Float16* xk16 = ws + 1 * NX;
  _Float16* xv16 = ws + 2 * NX;
  _Float16* Qh   = ws + 3 * NX;   // [b][h][s][dk]
  _Float16* Kh   = ws + 4 * NX;   // [b][h][s][dk]
  _Float16* Vt   = ws + 5 * NX;   // [b][h][dk][s]
  _Float16* ctx  = ws + 6 * NX;   // [b][s][d]
  _Float16* wq16 = ws + 7 * NX;
  _Float16* wk16 = wq16 + NW;
  _Float16* wv16 = wk16 + NW;
  _Float16* wo16 = wv16 + NW;
  float*    ml   = (float*)(wo16 + NW);   // [b][h][s] x {m,l}  (f32 pairs)

  float* out_y    = (float*)d_out;        // (B,S,D)
  float* attn_out = out_y + NX;           // (B,S,S)

  cvt_f32_f16<<<(int)(NX / 4 + 255) / 256, 256, 0, stream>>>(q,  xq16, (int)NX);
  cvt_f32_f16<<<(int)(NX / 4 + 255) / 256, 256, 0, stream>>>(k,  xk16, (int)NX);
  cvt_f32_f16<<<(int)(NX / 4 + 255) / 256, 256, 0, stream>>>(v,  xv16, (int)NX);
  cvt_f32_f16<<<(int)(NW / 4 + 255) / 256, 256, 0, stream>>>(Wq, wq16, (int)NW);
  cvt_f32_f16<<<(int)(NW / 4 + 255) / 256, 256, 0, stream>>>(Wk, wk16, (int)NW);
  cvt_f32_f16<<<(int)(NW / 4 + 255) / 256, 256, 0, stream>>>(Wv, wv16, (int)NW);
  cvt_f32_f16<<<(int)(NW / 4 + 255) / 256, 256, 0, stream>>>(Wo, wo16, (int)NW);

  dim3 pg(BATCH * SEQ / 16, D_MODEL / 64);            // (512, 8)
  proj_qkv_kernel<<<pg, 32, 0, stream>>>(xq16, wq16, bq, Qh, 0);
  proj_qkv_kernel<<<pg, 32, 0, stream>>>(xk16, wk16, bk, Kh, 0);
  proj_qkv_kernel<<<pg, 32, 0, stream>>>(xv16, wv16, bv, Vt, 1);

  dim3 fg(SEQ / 16, NHEAD, BATCH);                    // (128, 8, 4)
  flash_attn_kernel<<<fg, 32, 0, stream>>>(Qh, Kh, Vt, mask, ctx, ml);

  dim3 mg(SEQ / 32, SEQ / 16, BATCH);                 // (64, 128, 4)
  attn_mean_kernel<<<mg, 32, 0, stream>>>(Qh, Kh, mask, ml, attn_out);

  oproj_kernel<<<pg, 32, 0, stream>>>(ctx, wo16, bo, out_y);
}
